// SMultiHeadedAttention_89739046683500
// MI455X (gfx1250) — matrix operand order repaired and checked by reading, not compiled, not run on hardware
//
#include <hip/hip_runtime.h>
#include <hip/hip_bf16.h>

// ---------------------------------------------------------------------------
// Types for CDNA5 WMMA
// ---------------------------------------------------------------------------
typedef __attribute__((ext_vector_type(16))) __bf16 v16bf;
typedef __attribute__((ext_vector_type(2)))  __bf16 v2bf;
typedef __attribute__((ext_vector_type(8)))  float  v8f;

#define TILE_M 128
#define TILE_N 128
#define TILE_K 32
#define LDS_STRIDE 40   // bf16 elems per LDS row (32 + 8 pad, 80B, 16B aligned)

struct bf16x16_bits { uint4 lo, hi; };

// A-fragment (16x32 bf16): lane l<16 holds K {0..7} u {16..23}; l>=16 holds
// K {8..15} u {24..31}  (ISA 7.12.2 16-bit A-matrix table).
static __device__ __forceinline__ v16bf load_fragA(const __bf16* row, int lhalf) {
    bf16x16_bits t;
    t.lo = *(const uint4*)(row + lhalf * 8);
    t.hi = *(const uint4*)(row + 16 + lhalf * 8);
    return __builtin_bit_cast(v16bf, t);
}
// B-fragment (32x16 bf16): lane l holds column n=l%16, K {8h..8h+7} u
// {16+8h..16+8h+7}, h=l/16 (same order as A). LDS_B stored [n][k].
static __device__ __forceinline__ v16bf load_fragB(const __bf16* col, int lhalf) {
    bf16x16_bits t;
    t.lo = *(const uint4*)(col + lhalf * 8);
    t.hi = *(const uint4*)(col + 16 + lhalf * 8);
    return __builtin_bit_cast(v16bf, t);
}

// ---------------------------------------------------------------------------
// Generic strided batched GEMM:  C[z] = scale * A[z](MxK) * B[z](KxN) + bias
//   A(m,k): Ab + m*a_ms + k*a_ks   AM=0: a_ks==1 (row-major), AM=1: a_ms==1
//   B(k,n): Bb + k*b_ks + n*b_ns   BM=0: b_ns==1 (row-major), BM=1: b_ks==1
//   C(r,c): Cb + r*o_rs + c*o_cs
//   z0 = z >> ZSHIFT; z1 = z & ((1<<ZSHIFT)-1)
// M%128==0, K%32==0 required. BOUND_N handles ragged N (branchless loads).
// All strides fit in 32 bits. Double-buffered LDS, register-staged pipeline.
// ---------------------------------------------------------------------------
template<int AM, int BM, bool BOUND_N, int ZSHIFT>
__global__ __launch_bounds__(256)
void gemm_bf16_wmma(const float* __restrict__ Ag, const float* __restrict__ Bg,
                    float* __restrict__ Cg, const float* __restrict__ bias,
                    int M, int N, int K,
                    int a_ms, int a_ks, int a_bs0, int a_bs1,
                    int b_ks, int b_ns, int b_bs0, int b_bs1,
                    int o_rs, int o_cs, int o_bs0, int o_bs1,
                    float scale)
{
    __shared__ __bf16 sA[2][TILE_M * LDS_STRIDE];
    __shared__ __bf16 sB[2][TILE_N * LDS_STRIDE];

    const int z  = blockIdx.z;
    const int z0 = z >> ZSHIFT;
    const int z1 = z & ((1 << ZSHIFT) - 1);
    const float* Ab = Ag + z0 * a_bs0 + z1 * a_bs1;
    const float* Bb = Bg + z0 * b_bs0 + z1 * b_bs1;
    float*       Cb = Cg + z0 * o_bs0 + z1 * o_bs1;

    const int m0 = blockIdx.x * TILE_M;
    const int n0 = blockIdx.y * TILE_N;

    const int tid   = threadIdx.x;
    const int wid   = tid >> 5;
    const int lane  = tid & 31;
    const int wmo   = (wid & 1) * 64;   // 2 waves x 64 rows
    const int wno   = (wid >> 1) * 32;  // 4 waves x 32 cols
    const int lhalf = lane >> 4;
    const int lmod  = lane & 15;

    v8f acc[4][2] = {};
    float2 ar[8], br[8];                 // register staging (16+16 floats)

    // ---- stage: global -> registers (float2 / global_load_b64) ----
    auto loadA = [&](int kt) {
        const int k0 = kt * TILE_K;
        if constexpr (AM == 0) {         // row-major A: pairs along k
            #pragma unroll
            for (int i = 0; i < 8; ++i) {
                int idxp = tid + i * 256;
                int m  = idxp >> 4;
                int kp = (idxp & 15) * 2;
                ar[i] = *(const float2*)&Ab[(m0 + m) * a_ms + (k0 + kp)];
            }
        } else {                         // a_ms == 1 (transposed A): pairs along m
            #pragma unroll
            for (int i = 0; i < 8; ++i) {
                int idxp = tid + i * 256;
                int k  = idxp >> 6;
                int mp = (idxp & 63) * 2;
                ar[i] = *(const float2*)&Ab[(k0 + k) * a_ks + (m0 + mp)];
            }
        }
    };
    auto loadB = [&](int kt) {
        const int k0 = kt * TILE_K;
        if constexpr (BM == 0) {         // row-major B: pairs along n
            #pragma unroll
            for (int i = 0; i < 8; ++i) {
                int idxp = tid + i * 256;
                int k  = idxp >> 6;
                int np = (idxp & 63) * 2;
                if constexpr (BOUND_N) {
                    int nc = min(n0 + np, N - 2);
                    float2 v = *(const float2*)&Bb[(k0 + k) * b_ks + nc];
                    bool ok = (n0 + np) < N;
                    br[i].x = ok ? v.x : 0.f;
                    br[i].y = ok ? v.y : 0.f;
                } else {
                    br[i] = *(const float2*)&Bb[(k0 + k) * b_ks + (n0 + np)];
                }
            }
        } else {                         // b_ks == 1 (transposed B): pairs along k
            #pragma unroll
            for (int i = 0; i < 8; ++i) {
                int idxp = tid + i * 256;
                int n  = idxp >> 4;
                int kp = (idxp & 15) * 2;
                if constexpr (BOUND_N) {
                    int nc = min(n0 + n, N - 1);
                    float2 v = *(const float2*)&Bb[(k0 + kp) + nc * b_ns];
                    bool ok = (n0 + n) < N;
                    br[i].x = ok ? v.x : 0.f;
                    br[i].y = ok ? v.y : 0.f;
                } else {
                    br[i] = *(const float2*)&Bb[(k0 + kp) + (n0 + n) * b_ns];
                }
            }
        }
    };

    // ---- stage: registers -> bf16 LDS (hardware f32->bf16 converts) ----
    auto storeA = [&](int buf) {
        if constexpr (AM == 0) {
            #pragma unroll
            for (int i = 0; i < 8; ++i) {
                int idxp = tid + i * 256;
                int m  = idxp >> 4;
                int kp = (idxp & 15) * 2;
                v2bf h; h.x = (__bf16)ar[i].x; h.y = (__bf16)ar[i].y;
                *(v2bf*)&sA[buf][m * LDS_STRIDE + kp] = h;
            }
        } else {
            #pragma unroll
            for (int i = 0; i < 8; ++i) {
                int idxp = tid + i * 256;
                int k  = idxp >> 6;
                int mp = (idxp & 63) * 2;
                sA[buf][mp * LDS_STRIDE + k]       = (__bf16)ar[i].x;
                sA[buf][(mp + 1) * LDS_STRIDE + k] = (__bf16)ar[i].y;
            }
        }
    };
    auto storeB = [&](int buf) {
        if constexpr (BM == 0) {
            #pragma unroll
            for (int i = 0; i < 8; ++i) {
                int idxp = tid + i * 256;
                int k  = idxp >> 6;
                int np = (idxp & 63) * 2;
                sB[buf][np * LDS_STRIDE + k]       = (__bf16)br[i].x;
                sB[buf][(np + 1) * LDS_STRIDE + k] = (__bf16)br[i].y;
            }
        } else {
            #pragma unroll
            for (int i = 0; i < 8; ++i) {
                int idxp = tid + i * 256;
                int n  = idxp >> 4;
                int kp = (idxp & 15) * 2;
                v2bf h; h.x = (__bf16)br[i].x; h.y = (__bf16)br[i].y;
                *(v2bf*)&sB[buf][n * LDS_STRIDE + kp] = h;
            }
        }
    };

    // ---- compute: LDS fragments -> 8x v_wmma_f32_16x16x32_bf16 ----
    auto compute = [&](int buf) {
        v16bf af[4], bfr[2];
        #pragma unroll
        for (int i = 0; i < 4; ++i)
            af[i] = load_fragA(&sA[buf][(wmo + i * 16 + lmod) * LDS_STRIDE], lhalf);
        #pragma unroll
        for (int j = 0; j < 2; ++j)
            bfr[j] = load_fragB(&sB[buf][(wno + j * 16 + lmod) * LDS_STRIDE], lhalf);
        #pragma unroll
        for (int i = 0; i < 4; ++i)
            #pragma unroll
            for (int j = 0; j < 2; ++j)
                acc[i][j] = __builtin_amdgcn_wmma_f32_16x16x32_bf16(
                    false, af[i], false, bfr[j], (short)0, acc[i][j], false, false);
    };

    // ---- pipelined main loop: load(kt+1) overlaps compute(kt) ----
    const int ktiles = K / TILE_K;
    loadA(0); loadB(0);
    for (int kt = 0; kt < ktiles; ++kt) {
        const int buf = kt & 1;
        storeA(buf);
        storeB(buf);
        __syncthreads();
        if (kt + 1 < ktiles) {
            // gfx1250 global_prefetch_b8 two tiles ahead
            __builtin_prefetch(Ab + m0 * a_ms + (kt + 2) * TILE_K * a_ks, 0, 1);
            __builtin_prefetch(Bb + (kt + 2) * TILE_K * b_ks + n0 * b_ns, 0, 1);
            loadA(kt + 1);
            loadB(kt + 1);
        }
        compute(buf);
    }

    // ---- store C: lanes 0-15 -> rows base+v, lanes 16-31 -> rows base+8+v ----
    #pragma unroll
    for (int i = 0; i < 4; ++i) {
        int mbase = m0 + wmo + i * 16 + lhalf * 8;
        #pragma unroll
        for (int j = 0; j < 2; ++j) {
            int col = n0 + wno + j * 16 + lmod;
            if (!BOUND_N || col < N) {
                int off = mbase * o_rs + col * o_cs;
                #pragma unroll
                for (int v = 0; v < 8; ++v) {
                    float r = acc[i][j][v] * scale;
                    if (bias) r += bias[mbase + v];
                    Cb[off] = r;
                    off += o_rs;
                }
            }
        }
    }
}

// ---------------------------------------------------------------------------
// Column-normalize first 128 columns of each Gram matrix G[j] (1024x1024)
// into P[j] (1024x128).  grid.x = 16*128, block = 256.
// ---------------------------------------------------------------------------
__global__ __launch_bounds__(256)
void colnorm_kernel(const float* __restrict__ G, float* __restrict__ P)
{
    const int j = blockIdx.x >> 7;
    const int c = blockIdx.x & 127;
    const float* g = G + (size_t)j * (1024 * 1024) + c;   // column c, stride 1024

    float v[4];
    float ss = 0.f;
    #pragma unroll
    for (int i = 0; i < 4; ++i) {
        int r = threadIdx.x + i * 256;
        v[i] = g[(size_t)r * 1024];
        ss += v[i] * v[i];
    }
    #pragma unroll
    for (int off = 16; off > 0; off >>= 1)
        ss += __shfl_xor(ss, off, 32);

    __shared__ float red[8];
    const int wid  = threadIdx.x >> 5;
    const int lane = threadIdx.x & 31;
    if (lane == 0) red[wid] = ss;
    __syncthreads();
    float tot = red[0] + red[1] + red[2] + red[3] + red[4] + red[5] + red[6] + red[7];
    float inv = rsqrtf(tot + 1e-12f);

    #pragma unroll
    for (int i = 0; i < 4; ++i) {
        int r = threadIdx.x + i * 256;
        P[(size_t)j * (1024 * 128) + (size_t)r * 128 + c] = v[i] * inv;
    }
}

// ---------------------------------------------------------------------------
// Softmax over rows of length 128 (wave32: one wave per row, 4 elems/lane)
// ---------------------------------------------------------------------------
__global__ __launch_bounds__(256)
void softmax_rows128(float* __restrict__ s, int nrows)
{
    const int row = blockIdx.x * 8 + (threadIdx.x >> 5);
    if (row >= nrows) return;
    const int lane = threadIdx.x & 31;

    float4* p = (float4*)(s + (size_t)row * 128);
    float4 val = p[lane];

    float m = fmaxf(fmaxf(val.x, val.y), fmaxf(val.z, val.w));
    #pragma unroll
    for (int off = 16; off > 0; off >>= 1)
        m = fmaxf(m, __shfl_xor(m, off, 32));

    val.x = __expf(val.x - m);
    val.y = __expf(val.y - m);
    val.z = __expf(val.z - m);
    val.w = __expf(val.w - m);

    float sum = val.x + val.y + val.z + val.w;
    #pragma unroll
    for (int off = 16; off > 0; off >>= 1)
        sum += __shfl_xor(sum, off, 32);

    float inv = 1.f / sum;
    val.x *= inv; val.y *= inv; val.z *= inv; val.w *= inv;
    p[lane] = val;
}

// ---------------------------------------------------------------------------
// Host-side orchestration
// B=8, D=1024, H=16, dim=64, TOPK=128, SEQ_N=SEQ_M=1024
// ---------------------------------------------------------------------------
extern "C" void kernel_launch(void* const* d_in, const int* in_sizes, int n_in,
                              void* d_out, int out_size, void* d_ws, size_t ws_size,
                              hipStream_t stream)
{
    (void)in_sizes; (void)n_in; (void)out_size; (void)ws_size;

    const float* query = (const float*)d_in[0];
    const float* key_  = (const float*)d_in[1];
    const float* value = (const float*)d_in[2];
    const float* wq = (const float*)d_in[3];
    const float* bq = (const float*)d_in[4];
    const float* wk = (const float*)d_in[5];
    const float* bk = (const float*)d_in[6];
    const float* wv = (const float*)d_in[7];
    const float* bv = (const float*)d_in[8];
    const float* wm = (const float*)d_in[9];
    const float* bm = (const float*)d_in[10];
    float* out = (float*)d_out;

    float* ws = (float*)d_ws;
    const long long NBD = 8LL * 1024 * 1024;     // 8,388,608 floats (B*D*SEQ)
    float* q  = ws;                              // (8,1024,1024)
    float* kk = ws + NBD;                        // (8,1024,1024)
    float* vv = ws + 2 * NBD;                    // (8,1024,1024)
    float* x  = ws + 3 * NBD;                    // (8,1024,1024)
    float* nk = ws + 4 * NBD;                    // (8,1024,128)
    float* nv = nk + 8LL * 1024 * 128;           // (8,1024,128)
    float* P  = nv + 8LL * 1024 * 128;           // (16,1024,128)
    float* G  = P + 16LL * 1024 * 128;           // (16,1024,1024)
    float* sc = G;                               // scores alias Gram (16M floats)

    const dim3 blk(256);

    // 1-3) q/k/v projections: row-major A (weights), row-major B (activations)
    gemm_bf16_wmma<0, 0, false, 0><<<dim3(8, 8, 8), blk, 0, stream>>>(
        wq, query, q, bq, 1024, 1024, 1024,
        1024, 1, 0, 0, 1024, 1, 1048576, 0, 1024, 1, 1048576, 0, 1.0f);
    gemm_bf16_wmma<0, 0, false, 0><<<dim3(8, 8, 8), blk, 0, stream>>>(
        wk, key_, kk, bk, 1024, 1024, 1024,
        1024, 1, 0, 0, 1024, 1, 1048576, 0, 1024, 1, 1048576, 0, 1.0f);
    gemm_bf16_wmma<0, 0, false, 0><<<dim3(8, 8, 8), blk, 0, stream>>>(
        wv, value, vv, bv, 1024, 1024, 1024,
        1024, 1, 0, 0, 1024, 1, 1048576, 0, 1024, 1, 1048576, 0, 1.0f);

    // 4-5) Gram matrices: G[j] = KV[j]^T KV[j]  (transposed A path)
    gemm_bf16_wmma<1, 0, false, 0><<<dim3(8, 8, 8), blk, 0, stream>>>(
        kk, kk, G, nullptr, 1024, 1024, 1024,
        1, 1024, 1048576, 0, 1024, 1, 1048576, 0, 1024, 1, 1048576, 0, 1.0f);
    gemm_bf16_wmma<1, 0, false, 0><<<dim3(8, 8, 8), blk, 0, stream>>>(
        vv, vv, G + 8LL * 1048576, nullptr, 1024, 1024, 1024,
        1, 1024, 1048576, 0, 1024, 1, 1048576, 0, 1024, 1, 1048576, 0, 1.0f);

    // 6) P[j] = normalize(G[j][:, :128])
    colnorm_kernel<<<dim3(16 * 128), blk, 0, stream>>>(G, P);

    // 7-8) rank-128 projection: new_kv[b] = kv[b] @ P
    gemm_bf16_wmma<0, 0, false, 0><<<dim3(8, 1, 8), blk, 0, stream>>>(
        kk, P, nk, nullptr, 1024, 128, 1024,
        1024, 1, 1048576, 0, 128, 1, 131072, 0, 128, 1, 131072, 0, 1.0f);
    gemm_bf16_wmma<0, 0, false, 0><<<dim3(8, 1, 8), blk, 0, stream>>>(
        vv, P + 8LL * 131072, nv, nullptr, 1024, 128, 1024,
        1024, 1, 1048576, 0, 128, 1, 131072, 0, 128, 1, 131072, 0, 1.0f);

    // 9) scores[b,h] = (q_h^T k_h) / 8   M=1024(n) N=128(m) K=64(d), z=b*16+h
    gemm_bf16_wmma<1, 0, false, 4><<<dim3(8, 1, 128), blk, 0, stream>>>(
        q, nk, sc, nullptr, 1024, 128, 64,
        1, 16384, 1048576, 1024,    // A[n][d] = q[b, d*16+h, n]   (AM=1)
        2048, 1, 131072, 128,       // B[d][m] = nk[b, d*16+h, m]  (BM=0)
        128, 1, 2097152, 131072,    // C row-major per (b,h)
        0.125f);

    // 10) softmax over last dim (131072 rows of 128)
    softmax_rows128<<<dim3(131072 / 8), blk, 0, stream>>>(sc, 131072);

    // 11) x[b, d*16+h, n] = sum_m prob[n,m] * nv[b, d*16+h, m]
    //     M=1024(n) N=64(d) K=128(m) -> ragged N
    gemm_bf16_wmma<0, 1, true, 4><<<dim3(8, 1, 128), blk, 0, stream>>>(
        sc, nv, x, nullptr, 1024, 64, 128,
        128, 1, 2097152, 131072,    // A = prob, row-major (AM=0)
        1, 2048, 131072, 128,       // B[m][d] = nv[b, d*16+h, m]  (BM=1)
        1, 16384, 1048576, 1024,    // C[n][d] -> x[b, d*16+h, n]
        0.125f * 8.f);              // == 1.0f

    // 12) out = wm @ x + bm
    gemm_bf16_wmma<0, 0, false, 0><<<dim3(8, 8, 8), blk, 0, stream>>>(
        wm, x, out, bm, 1024, 1024, 1024,
        1024, 1, 0, 0, 1024, 1, 1048576, 0, 1024, 1, 1048576, 0, 1.0f);
}
